// AdversarialValidator_17918603559270
// MI455X (gfx1250) — matrix operand order repaired and checked by reading, not compiled, not run on hardware
//
#include <hip/hip_runtime.h>
#include <hip/hip_bf16.h>
#include <math.h>

// Problem constants (match reference)
#define EMB  512
#define HID  256
#define NH   8
#define DH   32
#define SEQ  256
#define BAT  8

typedef __bf16 bf16x16 __attribute__((ext_vector_type(16)));
typedef __bf16 bf16x4  __attribute__((ext_vector_type(4)));
typedef float  floatx8 __attribute__((ext_vector_type(8)));

// ---------------------------------------------------------------------------
// WMMA fragment loaders (wave32; element s of lane half h holds k = 16h + s, A and B alike)
// ---------------------------------------------------------------------------
__device__ __forceinline__ bf16x16 frag_A(const __bf16* sh, int rowBase, int lane) {
  int m = lane & 15, half = lane >> 4;
  const __bf16* p = sh + (rowBase + m) * 32;
  bf16x16 f;
#pragma unroll
  for (int j = 0; j < 8; ++j) {
    int kb = half * 16 + j * 2;
    f[2 * j]     = p[kb];
    f[2 * j + 1] = p[kb + 1];
  }
  return f;
}

// B tile stored as [N=64 rows][K=32] (W rows along K) -> transposed GEMM
__device__ __forceinline__ bf16x16 frag_BT(const __bf16* sh, int nBase, int lane) {
  int n = lane & 15, half = lane >> 4;
  const __bf16* p = sh + (nBase + n) * 32;
  bf16x16 f;
#pragma unroll
  for (int j = 0; j < 8; ++j) {
    int kb = half * 16 + j * 2;
    f[2 * j]     = p[kb];
    f[2 * j + 1] = p[kb + 1];
  }
  return f;
}

// B tile stored as [K=32 rows][N=64] (row-major K x N operand)
__device__ __forceinline__ bf16x16 frag_BN(const __bf16* sh, int nBase, int lane) {
  int n = (lane & 15) + nBase, half = lane >> 4;
  bf16x16 f;
#pragma unroll
  for (int j = 0; j < 8; ++j) {
    int kb = half * 16 + j * 2;
    f[2 * j]     = sh[kb * 64 + n];
    f[2 * j + 1] = sh[(kb + 1) * 64 + n];
  }
  return f;
}

__device__ __forceinline__ bf16x4 cvt4(float4 v) {
  bf16x4 o;
  o[0] = (__bf16)v.x; o[1] = (__bf16)v.y; o[2] = (__bf16)v.z; o[3] = (__bf16)v.w;
  return o;
}

// ---------------------------------------------------------------------------
// Generic batched bf16-WMMA GEMM, double-buffered LDS software pipeline:
//   TRANSB=1: C[m,n] = scale * sum_k A[m,k]*W[n,k] + bias[n]   (A @ W^T)
//   TRANSB=0: C[m,n] = scale * sum_k A[m,k]*Bm[k,n] + bias[n]  (A @ B)
// z batching: ptr += (z/zdiv)*off1 + (z%zdiv)*off2 for A, B, C independently.
// Block = 128 threads (4 waves); tile 64x64, K-step 32.
// GUARD=false requires M%64==0, N%64==0 (all launches except ctx GEMM, N=32).
// ---------------------------------------------------------------------------
template <bool TRANSB, bool GUARD>
__global__ __launch_bounds__(128)
void gemm_wmma(const float* __restrict__ A, const float* __restrict__ Bm,
               const float* __restrict__ bias, float* __restrict__ C,
               int M, int N, int K, int lda, int ldb, int ldc, float scale,
               int zdiv, long offA1, long offA2, long offB1, long offB2,
               long offC1, long offC2) {
  __shared__ __bf16 shA[2][64 * 32];
  __shared__ __bf16 shB[2][64 * 32];

  const int tid = threadIdx.x, lane = tid & 31, wave = tid >> 5;
  const int n0 = blockIdx.x * 64, m0 = blockIdx.y * 64;
  const int z = blockIdx.z;
  A  += (long)(z / zdiv) * offA1 + (long)(z % zdiv) * offA2;
  Bm += (long)(z / zdiv) * offB1 + (long)(z % zdiv) * offB2;
  C  += (long)(z / zdiv) * offC1 + (long)(z % zdiv) * offC2;

  // Phase helpers: tile = 64x32 = 512 float4 groups / 128 threads = 4 each.
  auto loadA = [&](int k0, float4* v) {
#pragma unroll
    for (int it = 0; it < 4; ++it) {
      int idx = it * 128 + tid, r = idx >> 3, kq = idx & 7;
      int gm = m0 + r;
      v[it] = make_float4(0.f, 0.f, 0.f, 0.f);
      if (!GUARD || gm < M)
        v[it] = *(const float4*)(A + (long)gm * lda + k0 + kq * 4);
    }
  };
  auto loadB = [&](int k0, float4* v) {
#pragma unroll
    for (int it = 0; it < 4; ++it) {
      int idx = it * 128 + tid;
      v[it] = make_float4(0.f, 0.f, 0.f, 0.f);
      if (TRANSB) {
        int r = idx >> 3, kq = idx & 7;
        int gn = n0 + r;
        if (!GUARD || gn < N)
          v[it] = *(const float4*)(Bm + (long)gn * ldb + k0 + kq * 4);
      } else {
        int kk = idx >> 4, nq = (idx & 15) * 4;
        int gn = n0 + nq;
        if (!GUARD || gn < N)  // group fully in/out (N % 4 == 0)
          v[it] = *(const float4*)(Bm + (long)(k0 + kk) * ldb + gn);
      }
    }
  };
  auto storeA = [&](int buf, const float4* v) {
#pragma unroll
    for (int it = 0; it < 4; ++it) {
      int idx = it * 128 + tid, r = idx >> 3, kq = idx & 7;
      *(bf16x4*)(&shA[buf][r * 32 + kq * 4]) = cvt4(v[it]);
    }
  };
  auto storeB = [&](int buf, const float4* v) {
#pragma unroll
    for (int it = 0; it < 4; ++it) {
      int idx = it * 128 + tid;
      if (TRANSB) {
        int r = idx >> 3, kq = idx & 7;
        *(bf16x4*)(&shB[buf][r * 32 + kq * 4]) = cvt4(v[it]);
      } else {
        int kk = idx >> 4, nq = (idx & 15) * 4;
        *(bf16x4*)(&shB[buf][kk * 64 + nq]) = cvt4(v[it]);
      }
    }
  };

  floatx8 acc[4];
#pragma unroll
  for (int c = 0; c < 4; ++c)
#pragma unroll
    for (int e = 0; e < 8; ++e) acc[c][e] = 0.f;

  float4 va[4], vb[4];
  loadA(0, va);
  loadB(0, vb);
  storeA(0, va);
  storeB(0, vb);
  __syncthreads();

  const int nk = K >> 5;
  for (int ki = 0; ki < nk; ++ki) {
    const int cur = ki & 1;
    const bool more = (ki + 1) < nk;

    // Issue next tile's global loads NOW; they stay in flight through the
    // ds_load + wmma block below (wait lands before the LDS store phase).
    if (more) {
      loadA((ki + 1) << 5, va);
      loadB((ki + 1) << 5, vb);
    }
    // Prefetch two K-steps ahead into GL2/WGP$.
    if (ki + 2 < nk) {
      int k2 = (ki + 2) << 5;
      int r = tid & 63, half = (tid >> 6) * 16;
      if (!GUARD || (m0 + r) < M)
        __builtin_prefetch(A + (long)(m0 + r) * lda + k2 + half, 0, 1);
      if (TRANSB) {
        if (!GUARD || (n0 + r) < N)
          __builtin_prefetch(Bm + (long)(n0 + r) * ldb + k2 + half, 0, 1);
      } else {
        int kk = tid >> 2, nq = (tid & 3) * 16;
        if (!GUARD || (n0 + nq) < N)
          __builtin_prefetch(Bm + (long)(k2 + kk) * ldb + n0 + nq, 0, 1);
      }
    }

    // Fragment loads (independent registers -> staggered dscnt waits),
    // then 4 back-to-back WMMAs.
    bf16x16 aF = frag_A(shA[cur], wave * 16, lane);
    bf16x16 bF[4];
#pragma unroll
    for (int c = 0; c < 4; ++c)
      bF[c] = TRANSB ? frag_BT(shB[cur], c * 16, lane)
                     : frag_BN(shB[cur], c * 16, lane);
#pragma unroll
    for (int c = 0; c < 4; ++c)
      acc[c] = __builtin_amdgcn_wmma_f32_16x16x32_bf16(
          false, aF, false, bF[c], (short)0, acc[c], false, false);

    if (more) {
      storeA(cur ^ 1, va);
      storeB(cur ^ 1, vb);
    }
    __syncthreads();
  }

  // epilogue: C/D layout — lane L: N = c*16 + L%16; VGPR r: M = (L/16)*8 + r
  const int mrow = m0 + wave * 16 + (lane >> 4) * 8;
  const int nc0  = n0 + (lane & 15);
#pragma unroll
  for (int c = 0; c < 4; ++c) {
    int n = nc0 + c * 16;
    if (GUARD && n >= N) continue;
    float bv = bias ? bias[n] : 0.f;
#pragma unroll
    for (int r = 0; r < 8; ++r) {
      int m = mrow + r;
      if (!GUARD || m < M) C[(long)m * ldc + n] = acc[c][r] * scale + bv;
    }
  }
}

// ---------------------------------------------------------------------------
// LayerNorm over HID then ReLU: one wave per row (HID=256 -> 8 per lane)
// ---------------------------------------------------------------------------
__global__ __launch_bounds__(256)
void ln_relu(const float* __restrict__ h, const float* __restrict__ g,
             const float* __restrict__ bb, float* __restrict__ out, int nrows) {
  int wave = threadIdx.x >> 5, lane = threadIdx.x & 31;
  int row = blockIdx.x * 8 + wave;
  if (row >= nrows) return;
  const float* r = h + (long)row * HID;
  float v[8], s = 0.f;
#pragma unroll
  for (int i = 0; i < 8; ++i) { v[i] = r[i * 32 + lane]; s += v[i]; }
#pragma unroll
  for (int o = 16; o > 0; o >>= 1) s += __shfl_xor(s, o, 32);
  float mu = s * (1.f / HID);
  float q = 0.f;
#pragma unroll
  for (int i = 0; i < 8; ++i) { float d = v[i] - mu; q += d * d; }
#pragma unroll
  for (int o = 16; o > 0; o >>= 1) q += __shfl_xor(q, o, 32);
  float rs = rsqrtf(q * (1.f / HID) + 1e-5f);
  float* w = out + (long)row * HID;
#pragma unroll
  for (int i = 0; i < 8; ++i) {
    int idx = i * 32 + lane;
    float y = (v[i] - mu) * rs * g[idx] + bb[idx];
    w[idx] = fmaxf(y, 0.f);
  }
}

// ---------------------------------------------------------------------------
// Row softmax over 256 columns, one wave per row (in place)
// ---------------------------------------------------------------------------
__global__ __launch_bounds__(256)
void softmax_rows(float* __restrict__ p) {
  int wave = threadIdx.x >> 5, lane = threadIdx.x & 31;
  long row = (long)blockIdx.x * 8 + wave;
  float* r = p + row * 256;
  float v[8], m = -INFINITY;
#pragma unroll
  for (int i = 0; i < 8; ++i) { v[i] = r[i * 32 + lane]; m = fmaxf(m, v[i]); }
#pragma unroll
  for (int o = 16; o > 0; o >>= 1) m = fmaxf(m, __shfl_xor(m, o, 32));
  float s = 0.f;
#pragma unroll
  for (int i = 0; i < 8; ++i) { v[i] = __expf(v[i] - m); s += v[i]; }
#pragma unroll
  for (int o = 16; o > 0; o >>= 1) s += __shfl_xor(s, o, 32);
  float inv = 1.f / s;
#pragma unroll
  for (int i = 0; i < 8; ++i) r[i * 32 + lane] = v[i] * inv;
}

// ---------------------------------------------------------------------------
// attn_weights[b,s,t] = mean over heads of attn
// ---------------------------------------------------------------------------
__global__ __launch_bounds__(256)
void attn_mean(const float* __restrict__ sc, float* __restrict__ out) {
  long idx = (long)blockIdx.x * 256 + threadIdx.x;  // B*S*S total
  int b = (int)(idx >> 16);
  int st = (int)(idx & 0xFFFF);
  float s = 0.f;
#pragma unroll
  for (int h = 0; h < NH; ++h) s += sc[((long)(b * NH + h) << 16) + st];
  out[idx] = s * (1.f / NH);
}

// ---------------------------------------------------------------------------
// pooled = mean_s attended; cons/hall 2-layer sigmoid heads. One block per b.
// ---------------------------------------------------------------------------
__global__ __launch_bounds__(256)
void pooled_heads(const float* __restrict__ att,
                  const float* __restrict__ cw1, const float* __restrict__ cb1,
                  const float* __restrict__ cw2, const float* __restrict__ cb2,
                  const float* __restrict__ hw1, const float* __restrict__ hb1,
                  const float* __restrict__ hw2, const float* __restrict__ hb2,
                  float* __restrict__ out) {
  __shared__ float sp[HID];
  __shared__ float rc[128], rh[128];
  int b = blockIdx.x, tid = threadIdx.x;
  float s = 0.f;
  for (int t = 0; t < SEQ; ++t) s += att[((long)b * SEQ + t) * HID + tid];
  sp[tid] = s * (1.f / SEQ);
  __syncthreads();
  if (tid < 128) {
    const float4* c4 = (const float4*)(cw1 + tid * HID);
    const float4* h4 = (const float4*)(hw1 + tid * HID);
    const float4* p4 = (const float4*)sp;
    float hc = cb1[tid], hh = hb1[tid];
    for (int j = 0; j < HID / 4; ++j) {
      float4 p = p4[j], a = c4[j], hb = h4[j];
      hc += p.x * a.x + p.y * a.y + p.z * a.z + p.w * a.w;
      hh += p.x * hb.x + p.y * hb.y + p.z * hb.z + p.w * hb.w;
    }
    rc[tid] = fmaxf(hc, 0.f) * cw2[tid];
    rh[tid] = fmaxf(hh, 0.f) * hw2[tid];
  }
  __syncthreads();
  for (int o = 64; o > 0; o >>= 1) {
    if (tid < o) { rc[tid] += rc[tid + o]; rh[tid] += rh[tid + o]; }
    __syncthreads();
  }
  if (tid == 0) {
    out[b]     = 1.f / (1.f + __expf(-(rc[0] + cb2[0])));
    out[8 + b] = 1.f / (1.f + __expf(-(rh[0] + hb2[0])));
  }
}

// ---------------------------------------------------------------------------
// Pairwise contradiction: for i<j:
//   sc = sigmoid( sum_t relu(a[b,i,t] + b[b,j,t] + cb1[t]) * cw2[t] + cb2 )
// M[b,i,j] = M[b,j,i] = sc ; diagonal = 0.
// Block = (b,i), thread j. b-rows staged through LDS (rows padded to 36
// floats); loads gathered into registers first so they overlap.
// ---------------------------------------------------------------------------
__global__ __launch_bounds__(256)
void pair_kernel(const float* __restrict__ am, const float* __restrict__ bm,
                 const float* __restrict__ cb1, const float* __restrict__ cw2,
                 const float* __restrict__ cb2, float* __restrict__ Mout) {
  __shared__ float arow[HID];
  __shared__ float w2[HID];
  __shared__ float bt[256 * 36];
  int b = blockIdx.y, i = blockIdx.x, tid = threadIdx.x;
  arow[tid] = am[((long)b * SEQ + i) * HID + tid] + cb1[tid];
  w2[tid] = cw2[tid];
  float acc = 0.f;
  const float4* brow4 = (const float4*)(bm + ((long)b * SEQ + tid) * HID);
  for (int t0 = 0; t0 < HID; t0 += 32) {
    float4 tmp[8];
#pragma unroll
    for (int t = 0; t < 8; ++t) tmp[t] = brow4[t0 / 4 + t];  // all in flight
    __syncthreads();
#pragma unroll
    for (int t = 0; t < 8; ++t) *(float4*)(bt + tid * 36 + t * 4) = tmp[t];
    __syncthreads();
    if (tid > i) {
#pragma unroll
      for (int t = 0; t < 32; ++t) {
        float v = arow[t0 + t] + bt[tid * 36 + t];
        acc += fmaxf(v, 0.f) * w2[t0 + t];
      }
    }
  }
  long base = (long)b * SEQ * SEQ;
  if (tid > i) {
    float sc = 1.f / (1.f + __expf(-(acc + cb2[0])));
    Mout[base + (long)i * SEQ + tid] = sc;
    Mout[base + (long)tid * SEQ + i] = sc;
  } else if (tid == i) {
    Mout[base + (long)i * SEQ + i] = 0.f;
  }
}

// ---------------------------------------------------------------------------
// Launcher. Workspace (floats):
//   h 524288 | rh 524288 | mf 524288 | qkv 1572864 | scores 4194304
//   ctx 524288 | attended 524288 | a 524288 | b 524288  => 36 MB total
// Output: cons[8] | hall[8] | M[8*256*256] | attn_weights[8*256*256]
// ---------------------------------------------------------------------------
extern "C" void kernel_launch(void* const* d_in, const int* in_sizes, int n_in,
                              void* d_out, int out_size, void* d_ws, size_t ws_size,
                              hipStream_t stream) {
  (void)in_sizes; (void)n_in; (void)out_size; (void)ws_size;
  const float* X       = (const float*)d_in[0];
  const float* enc_w1  = (const float*)d_in[1];
  const float* enc_b1  = (const float*)d_in[2];
  const float* ln_g    = (const float*)d_in[3];
  const float* ln_b    = (const float*)d_in[4];
  const float* enc_w2  = (const float*)d_in[5];
  const float* enc_b2  = (const float*)d_in[6];
  const float* ipw     = (const float*)d_in[7];
  const float* ipb     = (const float*)d_in[8];
  const float* out_w   = (const float*)d_in[9];
  const float* out_b   = (const float*)d_in[10];
  const float* cons_w1 = (const float*)d_in[11];
  const float* cons_b1 = (const float*)d_in[12];
  const float* cons_w2 = (const float*)d_in[13];
  const float* cons_b2 = (const float*)d_in[14];
  const float* hall_w1 = (const float*)d_in[15];
  const float* hall_b1 = (const float*)d_in[16];
  const float* hall_w2 = (const float*)d_in[17];
  const float* hall_b2 = (const float*)d_in[18];
  const float* con_w1  = (const float*)d_in[19];
  const float* con_b1  = (const float*)d_in[20];
  const float* con_w2  = (const float*)d_in[21];
  const float* con_b2  = (const float*)d_in[22];
  float* out = (float*)d_out;

  float* ws  = (float*)d_ws;
  float* h   = ws;
  float* rh  = h   + 524288;
  float* mf  = rh  + 524288;
  float* qkv = mf  + 524288;
  float* sc  = qkv + 1572864;
  float* ctx = sc  + 4194304;
  float* att = ctx + 524288;
  float* am  = att + 524288;
  float* bm  = am  + 524288;

  const int M0 = BAT * SEQ;             // 2048
  const long zA = (long)SEQ * 3 * HID;  // qkv batch stride (per b)
  const float iscale = 0.17677669529663689f;  // 1/sqrt(32)

  // 1. h = X @ enc_w1^T + b1
  gemm_wmma<true, false><<<dim3(HID / 64, M0 / 64, 1), 128, 0, stream>>>(
      X, enc_w1, enc_b1, h, M0, HID, EMB, EMB, EMB, HID, 1.f, 1, 0, 0, 0, 0, 0, 0);
  // 2. rh = relu(LN(h))
  ln_relu<<<M0 / 8, 256, 0, stream>>>(h, ln_g, ln_b, rh, M0);
  // 3. mf = rh @ enc_w2^T + b2
  gemm_wmma<true, false><<<dim3(HID / 64, M0 / 64, 1), 128, 0, stream>>>(
      rh, enc_w2, enc_b2, mf, M0, HID, HID, HID, HID, HID, 1.f, 1, 0, 0, 0, 0, 0, 0);
  // 4. qkv = mf @ in_proj_w^T + in_proj_b
  gemm_wmma<true, false><<<dim3(3 * HID / 64, M0 / 64, 1), 128, 0, stream>>>(
      mf, ipw, ipb, qkv, M0, 3 * HID, HID, HID, HID, 3 * HID, 1.f, 1, 0, 0, 0, 0, 0, 0);
  // 5/6. a = mf @ Wa^T ; b = mf @ Wb^T  (con_w1 split along columns, ldw=2*HID)
  gemm_wmma<true, false><<<dim3(HID / 64, M0 / 64, 1), 128, 0, stream>>>(
      mf, con_w1, nullptr, am, M0, HID, HID, HID, 2 * HID, HID, 1.f, 1, 0, 0, 0, 0, 0, 0);
  gemm_wmma<true, false><<<dim3(HID / 64, M0 / 64, 1), 128, 0, stream>>>(
      mf, con_w1 + HID, nullptr, bm, M0, HID, HID, HID, 2 * HID, HID, 1.f, 1, 0, 0, 0, 0, 0, 0);
  // 7. scores[b,h] = (q @ k^T) / sqrt(DH), batched over z = b*NH + h
  gemm_wmma<true, false><<<dim3(SEQ / 64, SEQ / 64, BAT * NH), 128, 0, stream>>>(
      qkv, qkv + HID, nullptr, sc, SEQ, SEQ, DH, 3 * HID, 3 * HID, SEQ, iscale,
      NH, zA, DH, zA, DH, (long)NH * SEQ * SEQ, (long)SEQ * SEQ);
  // 8. softmax rows
  softmax_rows<<<(BAT * NH * SEQ) / 8, 256, 0, stream>>>(sc);
  // 9. attn_weights = head mean
  attn_mean<<<(BAT * SEQ * SEQ) / 256, 256, 0, stream>>>(sc, out + 16 + BAT * SEQ * SEQ);
  // 10. ctx[b,h] = attn @ v (B operand row-major, N=32 -> guarded path)
  gemm_wmma<false, true><<<dim3(1, SEQ / 64, BAT * NH), 128, 0, stream>>>(
      sc, qkv + 2 * HID, nullptr, ctx, SEQ, DH, SEQ, SEQ, 3 * HID, HID, 1.f,
      NH, (long)NH * SEQ * SEQ, (long)SEQ * SEQ, zA, DH, (long)SEQ * HID, DH);
  // 11. attended = ctx @ out_w^T + out_b
  gemm_wmma<true, false><<<dim3(HID / 64, M0 / 64, 1), 128, 0, stream>>>(
      ctx, out_w, out_b, att, M0, HID, HID, HID, HID, HID, 1.f, 1, 0, 0, 0, 0, 0, 0);
  // 12. pooled + cons/hall heads -> out[0:16]
  pooled_heads<<<BAT, 256, 0, stream>>>(att, cons_w1, cons_b1, cons_w2, cons_b2,
                                        hall_w1, hall_b1, hall_w2, hall_b2, out);
  // 13. pairwise contradiction matrix -> out[16 : 16+B*S*S]
  pair_kernel<<<dim3(SEQ, BAT), 256, 0, stream>>>(am, bm, con_b1, con_w2, con_b2, out + 16);
}
